// SelfAttention_17179869902
// MI455X (gfx1250) — compile-verified, matrix operand order checked by reading, not run on hardware
//
#include <hip/hip_runtime.h>

// ---------------------------------------------------------------------------
// Self-attention (B=4, C=256, H=W=64 -> N=4096, Dqk=32) for gfx1250 (MI455X).
// All GEMMs on v_wmma_f32_16x16x32_f16 (wave32), f32 accumulation.
//   Stage 1: Q/K/V projections (fp32 x -> f16 tiles); V stored in a blocked
//            [b][mblock][c][32m] layout so attention staging is contiguous.
//   Stage 2: flash attention, 8 waves/workgroup = 4 n-tiles x 2 channel
//            halves covering 64 rows x 256 channels; K/V staged per 32-m
//            block into double-buffered LDS via async global->LDS copies
//            (ASYNCcnt), consumed through ds_load in WMMA operand layout.
// ---------------------------------------------------------------------------

typedef __attribute__((ext_vector_type(16))) _Float16 v16h;
typedef __attribute__((ext_vector_type(8)))  _Float16 v8h;
typedef __attribute__((ext_vector_type(4)))  _Float16 v4h;
typedef __attribute__((ext_vector_type(8)))  float    v8f;
typedef __attribute__((ext_vector_type(4)))  float    f32x4;
typedef __attribute__((ext_vector_type(4)))  int      i32x4;
typedef __attribute__((ext_vector_type(2)))  int      i32x2;

#define DEVI static __device__ __forceinline__

static constexpr int BB = 4;     // batch
static constexpr int CC = 256;   // channels
static constexpr int NN = 4096;  // H*W
static constexpr int DD = 32;    // Dqk
static constexpr int NMB = NN / 32;  // 128 m-blocks

// K-index packing for 16-bit A/B operands of V_WMMA_*_16X16X32 (wave32):
//   VGPR v in 0..3 holds K = 2v,2v+1   (+8 if lane>=16)
//   VGPR v in 4..7 holds K = 2v+8,2v+9 (+8 if lane>=16)
DEVI int kmap(int v, int half) { return (v < 4 ? 2 * v : 8 + 2 * v) + 8 * half; }

DEVI v8f wmma16(v16h a, v16h b, v8f c) {
  return __builtin_amdgcn_wmma_f32_16x16x32_f16(false, a, false, b, (short)0, c,
                                                false, false);
}

// ---- CDNA5 async global->LDS staging (guarded; falls back to sync copy) ----
#if defined(__has_builtin)
#if __has_builtin(__builtin_amdgcn_global_load_async_to_lds_b128) && \
    __has_builtin(__builtin_amdgcn_global_load_async_to_lds_b64)
#define HAVE_ASYNC_LDS 1
#endif
#endif

DEVI void cp128(const _Float16* g, _Float16* l) {
#if defined(HAVE_ASYNC_LDS)
  __builtin_amdgcn_global_load_async_to_lds_b128(
      (__attribute__((address_space(1))) i32x4*)g,
      (__attribute__((address_space(3))) i32x4*)l, 0, 0);
#else
  *(v8h*)l = *(const v8h*)g;
#endif
}

DEVI void cp64(const _Float16* g, _Float16* l) {
#if defined(HAVE_ASYNC_LDS)
  __builtin_amdgcn_global_load_async_to_lds_b64(
      (__attribute__((address_space(1))) i32x2*)g,
      (__attribute__((address_space(3))) i32x2*)l, 0, 0);
#else
  *(v4h*)l = *(const v4h*)g;
#endif
}

DEVI void wait_async_prev() {  // drain previous block's 5 async ops
#if defined(HAVE_ASYNC_LDS)
#if __has_builtin(__builtin_amdgcn_s_wait_asynccnt)
  __builtin_amdgcn_s_wait_asynccnt(5);
#else
  asm volatile("s_wait_asynccnt 0x5" ::: "memory");
#endif
#endif
}

DEVI void wait_async_all() {
#if defined(HAVE_ASYNC_LDS)
#if __has_builtin(__builtin_amdgcn_s_wait_asynccnt)
  __builtin_amdgcn_s_wait_asynccnt(0);
#else
  asm volatile("s_wait_asynccnt 0x0" ::: "memory");
#endif
#endif
}

// ---------------------------------------------------------------------------
// Stage 1: projections.  grid = 256 blocks x 128 threads (4 waves).
// Each wave owns one (b, 16-row n-tile): A = x^T tile (16n x 256c) in regs.
// Qh,Kh: [B][N][32] f16.  Vb (blocked): [B][N/32][C][32] f16.
// ---------------------------------------------------------------------------
__global__ __launch_bounds__(128) void proj_kernel(
    const float* __restrict__ x,
    const float* __restrict__ Wq, const float* __restrict__ bq,
    const float* __restrict__ Wk, const float* __restrict__ bk,
    const float* __restrict__ Wv, const float* __restrict__ bv,
    _Float16* __restrict__ Qh, _Float16* __restrict__ Kh,
    _Float16* __restrict__ Vb) {
  const int lane = threadIdx.x & 31;
  const int wave = threadIdx.x >> 5;
  const int tile = blockIdx.x * 4 + wave;   // 0..1023
  const int b    = tile >> 8;
  const int n0   = (tile & 255) << 4;
  const int half = lane >> 4;
  const int r    = lane & 15;
  const int n    = n0 + r;

  const float* xb = x + (size_t)b * CC * NN;

  // A blocks: x^T (rows n = lane, K = channel c), 8 blocks of 32 channels.
  v16h A[8];
#pragma unroll
  for (int kb = 0; kb < 8; ++kb) {
#pragma unroll
    for (int v = 0; v < 8; ++v) {
      const int c = kb * 32 + kmap(v, half);
      A[kb][2 * v]     = (_Float16)xb[(size_t)c * NN + n];
      A[kb][2 * v + 1] = (_Float16)xb[(size_t)(c + 1) * NN + n];
    }
  }

  auto loadB = [&](const float* Wmat, int dcol0, int kb) -> v16h {
    v16h bt;
#pragma unroll
    for (int v = 0; v < 8; ++v) {
      const int c = kb * 32 + kmap(v, half);
      const float* p = Wmat + (size_t)(dcol0 + r) * CC + c;
      bt[2 * v]     = (_Float16)p[0];
      bt[2 * v + 1] = (_Float16)p[1];
    }
    return bt;
  };

  // ---- Q and K (2 column tiles each) ----
#pragma unroll
  for (int ot = 0; ot < 2; ++ot) {
    v8f accq = {};
    v8f acck = {};
#pragma unroll
    for (int kb = 0; kb < 8; ++kb) {
      accq = wmma16(A[kb], loadB(Wq, ot * 16, kb), accq);
      acck = wmma16(A[kb], loadB(Wk, ot * 16, kb), acck);
    }
    const int d = ot * 16 + r;
    const float biasq = bq[d];
    const float biask = bk[d];
#pragma unroll
    for (int j = 0; j < 8; ++j) {
      const int nn = n0 + j + 8 * half;
      Qh[((size_t)b * NN + nn) * DD + d] = (_Float16)(accq[j] + biasq);
      Kh[((size_t)b * NN + nn) * DD + d] = (_Float16)(acck[j] + biask);
    }
  }

  // ---- V (16 column tiles) into blocked [b][nblk][c][32] layout ----
  const int nblk = n0 >> 5;          // 32-row block index
  const int nw   = (n0 & 31);        // 0 or 16 within block
  for (int ot = 0; ot < 16; ++ot) {
    v8f acc = {};
#pragma unroll
    for (int kb = 0; kb < 8; ++kb)
      acc = wmma16(A[kb], loadB(Wv, ot * 16, kb), acc);
    const int c = ot * 16 + r;
    const float bias = bv[c];
    v8h o;
#pragma unroll
    for (int j = 0; j < 8; ++j) o[j] = (_Float16)(acc[j] + bias);
    // rows j contiguous -> one 16B store per lane
    const size_t idx =
        ((((size_t)b * NMB + nblk) * CC + c) * 32) + nw + 8 * half;
    *(v8h*)(Vb + idx) = o;
  }
}

// ---------------------------------------------------------------------------
// Stage 2: flash attention.
// grid = 256 blocks (one per (b, 64-row slab)) x 256 threads = 8 waves.
// Wave w: n-tile = w>>1 (4 tiles of 16 rows), channel half = w&1 (128 ch).
// Per 32-m block: K (2KB) + V (16KB) staged into double-buffered LDS with
// async copies (5 ops/thread/block), then 2 S^T WMMAs + online softmax +
// 8 V*P WMMAs per wave, operands read from LDS via ds_load.
// ---------------------------------------------------------------------------
__global__ __launch_bounds__(256) void attn_kernel(
    const _Float16* __restrict__ Qh, const _Float16* __restrict__ Kh,
    const _Float16* __restrict__ Vb, const float* __restrict__ x,
    const float* __restrict__ gamma, float* __restrict__ out) {
  __shared__ alignas(16) _Float16 sK[2][32 * 32];    //  4 KB
  __shared__ alignas(16) _Float16 sV[2][CC * 32];    // 32 KB

  const int tid  = threadIdx.x;
  const int lane = tid & 31;
  const int wave = tid >> 5;
  const int b    = blockIdx.x >> 6;              // 4 batches x 64 slabs
  const int nbase = (blockIdx.x & 63) << 6;      // 64-row slab
  const int half = lane >> 4;
  const int r    = lane & 15;
  const int n0   = nbase + (wave >> 1) * 16;     // this wave's n-tile
  const int c0   = (wave & 1) * 128;             // this wave's channel half
  const float scale = 0.17677669529663687f;      // 1/sqrt(32)

  const _Float16* kglob = Kh + (size_t)b * NN * DD;           // [m][32]
  const _Float16* vglob = Vb + (size_t)b * NMB * CC * 32;     // [mb][c][32]

  // stage one 32-m block into LDS buffer `buf` (5 async ops per thread)
  auto stage = [&](int mb, int buf) {
    // K: 2KB contiguous; 256 threads x 8B
    cp64(kglob + (size_t)mb * 32 * DD + tid * 4, &sK[buf][tid * 4]);
    // V: 16KB contiguous; 256 threads x 4 x 16B
    const _Float16* vsrc = vglob + (size_t)mb * CC * 32;
#pragma unroll
    for (int i = 0; i < 4; ++i) {
      const int o = (tid * 4 + i) * 8;
      cp128(vsrc + o, &sV[buf][o]);
    }
  };

  // Q as B-matrix (32d x 16n), register-resident for the whole kernel.
  v16h qb;
  {
    const _Float16* qp = Qh + ((size_t)b * NN + n0 + r) * DD;
#pragma unroll
    for (int v = 0; v < 8; ++v) {
      const int d = kmap(v, half);
      qb[2 * v]     = qp[d];
      qb[2 * v + 1] = qp[d + 1];
    }
  }

  v8f acc[8] = {{}, {}, {}, {}, {}, {}, {}, {}};
  float mrow = -3.0e38f;
  float ssum = 0.0f;

  stage(0, 0);

  for (int mb = 0; mb < NMB; ++mb) {
    const int cur = mb & 1;
    if (mb + 1 < NMB) {
      stage(mb + 1, cur ^ 1);
      wait_async_prev();           // drain block mb's 5 ops
    } else {
      wait_async_all();
    }
    __syncthreads();               // LDS buffer `cur` visible to all waves

    const _Float16* sk = sK[cur];
    const _Float16* sv = sV[cur];

    // K A-tiles from LDS (rows m = lane, K = d)
    v16h ka0, ka1;
#pragma unroll
    for (int v = 0; v < 8; ++v) {
      const int d = kmap(v, half);
      ka0[2 * v] = sk[r * 32 + d];        ka0[2 * v + 1] = sk[r * 32 + d + 1];
      ka1[2 * v] = sk[(16 + r) * 32 + d]; ka1[2 * v + 1] = sk[(16 + r) * 32 + d + 1];
    }

    // S^T tiles: D[row m = j + 8*half][col n = r]
    const v8f z = {};
    v8f s0 = wmma16(ka0, qb, z);
    v8f s1 = wmma16(ka1, qb, z);

    float pmax = -3.0e38f;
#pragma unroll
    for (int j = 0; j < 8; ++j) {
      s0[j] = fminf(10.f, fmaxf(-10.f, s0[j] * scale));
      s1[j] = fminf(10.f, fmaxf(-10.f, s1[j] * scale));
      pmax  = fmaxf(pmax, fmaxf(s0[j], s1[j]));
    }
    pmax = fmaxf(pmax, __shfl_xor(pmax, 16));
    const float mnew = fmaxf(mrow, pmax);
    const float corr = __expf(mrow - mnew);

    float e0[8], e1[8], esum = 0.f;
#pragma unroll
    for (int j = 0; j < 8; ++j) {
      e0[j] = __expf(s0[j] - mnew);
      e1[j] = __expf(s1[j] - mnew);
      esum += e0[j] + e1[j];
    }
    esum += __shfl_xor(esum, 16);
    ssum = ssum * corr + esum;
    mrow = mnew;

    // exp(S) -> f16 A matrix; D layout maps 1:1 onto the A K-packing.
    v16h ap;
#pragma unroll
    for (int v = 0; v < 4; ++v) {
      ap[2 * v]         = (_Float16)e0[2 * v];
      ap[2 * v + 1]     = (_Float16)e0[2 * v + 1];
      ap[8 + 2 * v]     = (_Float16)e1[2 * v];
      ap[8 + 2 * v + 1] = (_Float16)e1[2 * v + 1];
    }

    // O += P * V^T for this wave's 8 channel tiles (V from LDS, c-major).
#pragma unroll
    for (int ct = 0; ct < 8; ++ct) {
      const int cl = c0 + ct * 16 + r;
      v16h vb;
#pragma unroll
      for (int v = 0; v < 8; ++v) {
        const int mm = kmap(v, half);
        vb[2 * v] = sv[cl * 32 + mm]; vb[2 * v + 1] = sv[cl * 32 + mm + 1];
      }
#pragma unroll
      for (int j = 0; j < 8; ++j) acc[ct][j] *= corr;
      acc[ct] = wmma16(ap, vb, acc[ct]);
    }

    __syncthreads();   // all waves done with buffer `cur` before re-staging
  }

  // Epilogue: out = gamma * (acc / ssum) + x   (contiguous float4 I/O).
  const float inv = 1.0f / ssum;
  const float g   = gamma[0];
#pragma unroll
  for (int ct = 0; ct < 8; ++ct) {
    const int c = c0 + ct * 16 + r;
    const size_t base = ((size_t)b * CC + c) * NN + n0 + 8 * half;
    const f32x4 x0 = *(const f32x4*)(x + base);
    const f32x4 x1 = *(const f32x4*)(x + base + 4);
    f32x4 o0, o1;
#pragma unroll
    for (int j = 0; j < 4; ++j) {
      o0[j] = g * (acc[ct][j] * inv) + x0[j];
      o1[j] = g * (acc[ct][4 + j] * inv) + x1[j];
    }
    *(f32x4*)(out + base)     = o0;
    *(f32x4*)(out + base + 4) = o1;
  }
}

// ---------------------------------------------------------------------------
extern "C" void kernel_launch(void* const* d_in, const int* in_sizes, int n_in,
                              void* d_out, int out_size, void* d_ws,
                              size_t ws_size, hipStream_t stream) {
  const float* x     = (const float*)d_in[0];
  const float* Wq    = (const float*)d_in[1];
  const float* bq    = (const float*)d_in[2];
  const float* Wk    = (const float*)d_in[3];
  const float* bk    = (const float*)d_in[4];
  const float* Wv    = (const float*)d_in[5];
  const float* bv    = (const float*)d_in[6];
  const float* gamma = (const float*)d_in[7];

  // Workspace: Qh (1 MB) | Kh (1 MB) | Vb blocked (8 MB), all f16.
  _Float16* Qh = (_Float16*)d_ws;
  _Float16* Kh = Qh + (size_t)BB * NN * DD;
  _Float16* Vb = Kh + (size_t)BB * NN * DD;

  proj_kernel<<<dim3(256), dim3(128), 0, stream>>>(x, Wq, bq, Wk, bk, Wv, bv,
                                                   Qh, Kh, Vb);
  attn_kernel<<<dim3(256), dim3(256), 0, stream>>>(Qh, Kh, Vb, x, gamma,
                                                   (float*)d_out);
}
